// MMD_loss_8581344657570
// MI455X (gfx1250) — matrix operand order repaired and checked by reading, not compiled, not run on hardware
//
#include <hip/hip_runtime.h>

#define NROWS 4096
#define DIM   512
#define NTOT  8192

typedef float v2f __attribute__((ext_vector_type(2)));
typedef float v8f __attribute__((ext_vector_type(8)));
typedef int   v4i __attribute__((ext_vector_type(4)));
typedef __attribute__((ext_vector_type(16))) __bf16 v16bf;

#define AS1 __attribute__((address_space(1)))
#define AS3 __attribute__((address_space(3)))

#if defined(__has_builtin)
#  if __has_builtin(__builtin_amdgcn_global_load_async_to_lds_b128)
#    define HAVE_ASYNC_LDS 1
#  endif
#endif
#ifndef HAVE_ASYNC_LDS
#  define HAVE_ASYNC_LDS 0
#endif

// workspace layout:
// float idx [0..1]  double accumulator
// float idx [2]     sumsq
// float idx [3..7]  nib[5] = -1/(bw*2^i)
// float idx [8 .. 8+8192)      row squared norms
// float idx [8200 .. 8712)     column sums
// byte 40960:  Xhi bf16 [8192][512]     (optional fast path)
// byte 40960 + 8MiB: Xlo bf16 [8192][512]
#define WS_ACC   0
#define WS_SUMSQ 2
#define WS_NIB   3
#define WS_SQ    8
#define WS_COL   8200
#define WS_BF16_BYTE_OFF 40960

#define BM 128
#define BN 128
#define BK 32
#define LDSW 36   // f32 fallback LDS stride (floats)
#define LDSB 40   // bf16 LDS stride (halves) = 80B: 16B-aligned, bank-spread
#define TILE_HALVES (BM * LDSB)   // one 128x32 bf16 tile in LDS (padded)

__device__ __forceinline__ const float* row_ptr(const float* src, const float* tgt, int i) {
    return (i < NROWS) ? (src + (size_t)i * DIM) : (tgt + (size_t)(i - NROWS) * DIM);
}

__device__ __forceinline__ unsigned short f2bf_rne(float f) {
    unsigned int u = __float_as_uint(f);
    unsigned int r = (u + 0x7fffu + ((u >> 16) & 1u)) >> 16;
    return (unsigned short)r;
}
__device__ __forceinline__ float bf2f(unsigned short h) {
    return __uint_as_float(((unsigned int)h) << 16);
}
__device__ __forceinline__ unsigned int pk(unsigned short a, unsigned short b) {
    return (unsigned int)a | ((unsigned int)b << 16);
}

__device__ __forceinline__ void wait_async_lds() {
#if HAVE_ASYNC_LDS
#  if __has_builtin(__builtin_amdgcn_s_wait_asynccnt)
    __builtin_amdgcn_s_wait_asynccnt(0);
#  else
    asm volatile("s_wait_asynccnt 0x0" ::: "memory");
#  endif
#endif
}

union FragBF { uint4 q[2]; v16bf v; };

// A 16x32 bf16 fragment: lane-half h -> K[8h..8h+8) in V0-3, K[16+8h..) in V4-7
__device__ __forceinline__ v16bf load_frag_a(const unsigned short* rowbase, int h) {
    FragBF f;
    f.q[0] = *(const uint4*)(rowbase + h * 8);
    f.q[1] = *(const uint4*)(rowbase + 16 + h * 8);
    return f.v;
}
// B 32x16 bf16 fragment: lane-half h -> K[8h..8h+8) in V0-3, K[16+8h..) in V4-7
__device__ __forceinline__ v16bf load_frag_b(const unsigned short* rowbase, int h) {
    FragBF f;
    f.q[0] = *(const uint4*)(rowbase + h * 8);
    f.q[1] = *(const uint4*)(rowbase + 16 + h * 8);
    return f.v;
}

__global__ void k_zero(float* __restrict__ ws, float* __restrict__ out) {
    int t = threadIdx.x;
    if (t < 512) ws[WS_COL + t] = 0.0f;
    if (t < 3)   ws[t] = 0.0f;
    if (t == 0)  out[0] = 0.0f;
}

// 32 rows per 256-thread block: row squared-norms, global sumsq, column sums.
__global__ __launch_bounds__(256) void k_rowstats(const float* __restrict__ src,
                                                  const float* __restrict__ tgt,
                                                  float* __restrict__ ws) {
    __shared__ float colacc[DIM];
    __shared__ float rowred[32][8];
    const int t = threadIdx.x;
    for (int c = t; c < DIM; c += 256) colacc[c] = 0.0f;
    __syncthreads();

    const int rloc = t >> 3;
    const int seg  = t & 7;
    const int row  = blockIdx.x * 32 + rloc;
    const float* p = row_ptr(src, tgt, row) + seg * 64;

    float ss = 0.0f;
#pragma unroll
    for (int i = 0; i < 16; ++i) {
        float4 x = ((const float4*)p)[i];
        ss += x.x * x.x + x.y * x.y + x.z * x.z + x.w * x.w;
        const int c = seg * 64 + i * 4;
        atomicAdd(&colacc[c + 0], x.x);
        atomicAdd(&colacc[c + 1], x.y);
        atomicAdd(&colacc[c + 2], x.z);
        atomicAdd(&colacc[c + 3], x.w);
    }
    rowred[rloc][seg] = ss;
    __syncthreads();

    if (t < 32) {
        float s = 0.0f;
#pragma unroll
        for (int k = 0; k < 8; ++k) s += rowred[t][k];
        ws[WS_SQ + blockIdx.x * 32 + t] = s;
        atomicAdd(&ws[WS_SUMSQ], s);
    }
    for (int c = t; c < DIM; c += 256) atomicAdd(&ws[WS_COL + c], colacc[c]);
}

// sum(l2) = 2N*sumsq - 2*||colsum||^2 ; derive 5 inverse-bandwidth factors.
__global__ void k_finalize(float* __restrict__ ws) {
    __shared__ float red[256];
    const int t = threadIdx.x;
    float c0 = ws[WS_COL + t];
    float c1 = ws[WS_COL + 256 + t];
    red[t] = c0 * c0 + c1 * c1;
    __syncthreads();
    for (int s = 128; s > 0; s >>= 1) {
        if (t < s) red[t] += red[t + s];
        __syncthreads();
    }
    if (t == 0) {
        double ssq   = (double)red[0];
        double sumsq = (double)ws[WS_SUMSQ];
        double N     = (double)NTOT;
        double suml2 = 2.0 * N * sumsq - 2.0 * ssq;
        double bw    = suml2 / (N * N - N) / 4.0;  // kernel_mul^(kernel_num//2) = 4
#pragma unroll
        for (int i = 0; i < 5; ++i)
            ws[WS_NIB + i] = (float)(-1.0 / (bw * (double)(1 << i)));
    }
}

// One-time f32 -> (bf16 hi, bf16 lo) split: x ~= hi + lo with ~16 mantissa bits.
__global__ __launch_bounds__(256) void k_convert(const float* __restrict__ src,
                                                 const float* __restrict__ tgt,
                                                 unsigned short* __restrict__ xhi,
                                                 unsigned short* __restrict__ xlo) {
    const int idx = (blockIdx.x * 256 + threadIdx.x) * 8;  // 8 elements/thread
    const int row = idx >> 9;
    const int col = idx & (DIM - 1);
    const float* p = row_ptr(src, tgt, row) + col;
    float4 x0 = ((const float4*)p)[0];
    float4 x1 = ((const float4*)p)[1];
    float xs[8] = {x0.x, x0.y, x0.z, x0.w, x1.x, x1.y, x1.z, x1.w};
    unsigned short hs[8], ls[8];
#pragma unroll
    for (int e = 0; e < 8; ++e) {
        unsigned short h = f2bf_rne(xs[e]);
        hs[e] = h;
        ls[e] = f2bf_rne(xs[e] - bf2f(h));
    }
    uint4 hv, lv;
    hv.x = pk(hs[0], hs[1]); hv.y = pk(hs[2], hs[3]);
    hv.z = pk(hs[4], hs[5]); hv.w = pk(hs[6], hs[7]);
    lv.x = pk(ls[0], ls[1]); lv.y = pk(ls[2], ls[3]);
    lv.z = pk(ls[4], ls[5]); lv.w = pk(ls[6], ls[7]);
    *(uint4*)(xhi + idx) = hv;
    *(uint4*)(xlo + idx) = lv;
}

// ---------------- Primary: split-bf16 WMMA (V_WMMA_F32_16X16X32_BF16) ----------------
__global__ __launch_bounds__(256) void k_mmd_bf16(const unsigned short* __restrict__ xhi,
                                                  const unsigned short* __restrict__ xlo,
                                                  float* __restrict__ ws) {
    // tiles 0..3: m-hi, m-lo, n-hi, n-lo (each 128x32 bf16, padded stride LDSB)
    __shared__ unsigned short lds[4 * TILE_HALVES];
    __shared__ float sqm[BM];
    __shared__ float sqn[BN];
    __shared__ float red[256];

    const int t    = threadIdx.x;
    const int lane = t & 31;
    const int wid  = t >> 5;
    const int wm   = wid >> 2;          // 0..1 -> 64-row slab
    const int wn   = wid & 3;           // 0..3 -> 32-col slab
    const int bi   = blockIdx.x;
    const int bj   = blockIdx.y;
    const int m0   = bi * BM;
    const int n0   = bj * BN;
    const int rl   = lane & 15;
    const int h    = lane >> 4;
    const int mh   = h << 3;            // C/D lane-half -> M offset

    if (t < BM) sqm[t] = ws[WS_SQ + m0 + t];
    else        sqn[t - BM] = ws[WS_SQ + n0 + (t - BM)];

    v8f acc[4][2];
    const v8f vzero = {0.f, 0.f, 0.f, 0.f, 0.f, 0.f, 0.f, 0.f};
#pragma unroll
    for (int mt = 0; mt < 4; ++mt)
#pragma unroll
        for (int nt = 0; nt < 2; ++nt) acc[mt][nt] = vzero;

    for (int k0 = 0; k0 < DIM; k0 += BK) {
        // stage 4 tiles of 128x32 bf16: 2048 x B128 chunks, 8 per thread
#pragma unroll
        for (int q = 0; q < 8; ++q) {
            const int s    = t + 256 * q;
            const int tile = s >> 9;                       // 0..3
            const int r    = (s >> 2) & 127;
            const int c    = (s & 3) << 3;                 // ushort offset 0,8,16,24
            // source row base: tiles 0/1 -> m-slab, 2/3 -> n-slab; even -> hi, odd -> lo
            const int grow = ((tile & 2) ? n0 : m0) + r;
            const unsigned short* gb = (tile & 1) ? xlo : xhi;
            const unsigned short* g  = gb + (size_t)grow * DIM + k0 + c;
            unsigned short*       l  = lds + tile * TILE_HALVES + r * LDSB + c;
#if HAVE_ASYNC_LDS
            __builtin_amdgcn_global_load_async_to_lds_b128((AS1 v4i*)g, (AS3 v4i*)l, 0, 0);
#else
            *(uint4*)l = *(const uint4*)g;
#endif
        }
        wait_async_lds();
        __syncthreads();

        v16bf bh[2], bl[2];
#pragma unroll
        for (int nt = 0; nt < 2; ++nt) {
            const int nr = (wn * 32 + nt * 16 + rl) * LDSB;
            bh[nt] = load_frag_b(lds + 2 * TILE_HALVES + nr, h);
            bl[nt] = load_frag_b(lds + 3 * TILE_HALVES + nr, h);
        }
#pragma unroll
        for (int mt = 0; mt < 4; ++mt) {
            const int mr = (wm * 64 + mt * 16 + rl) * LDSB;
            const v16bf ah = load_frag_a(lds + 0 * TILE_HALVES + mr, h);
            const v16bf al = load_frag_a(lds + 1 * TILE_HALVES + mr, h);
#pragma unroll
            for (int nt = 0; nt < 2; ++nt) {
                acc[mt][nt] = __builtin_amdgcn_wmma_f32_16x16x32_bf16(
                    false, ah, false, bh[nt], (short)0, acc[mt][nt], false, false);
                acc[mt][nt] = __builtin_amdgcn_wmma_f32_16x16x32_bf16(
                    false, ah, false, bl[nt], (short)0, acc[mt][nt], false, false);
                acc[mt][nt] = __builtin_amdgcn_wmma_f32_16x16x32_bf16(
                    false, al, false, bh[nt], (short)0, acc[mt][nt], false, false);
            }
        }
        __syncthreads();
    }

    const float nb0 = ws[WS_NIB + 0], nb1 = ws[WS_NIB + 1], nb2 = ws[WS_NIB + 2],
                nb3 = ws[WS_NIB + 3], nb4 = ws[WS_NIB + 4];

    float lsum = 0.0f;
#pragma unroll
    for (int mt = 0; mt < 4; ++mt) {
#pragma unroll
        for (int nt = 0; nt < 2; ++nt) {
            const float sj = sqn[wn * 32 + nt * 16 + rl];
#pragma unroll
            for (int v = 0; v < 8; ++v) {
                const float si = sqm[wm * 64 + mt * 16 + mh + v];
                float l2 = si + sj - 2.0f * acc[mt][nt][v];
                l2 = fmaxf(l2, 0.0f);
                lsum += __expf(l2 * nb0) + __expf(l2 * nb1) + __expf(l2 * nb2)
                      + __expf(l2 * nb3) + __expf(l2 * nb4);
            }
        }
    }

    red[t] = lsum;
    __syncthreads();
    for (int s = 128; s > 0; s >>= 1) {
        if (t < s) red[t] += red[t + s];
        __syncthreads();
    }
    if (t == 0) {
        const float sgn = ((bi < (NROWS / BM)) == (bj < (NROWS / BN))) ? 1.0f : -1.0f;
        atomicAdd((double*)&ws[WS_ACC], (double)(sgn * red[0]));
    }
}

// ---------------- Fallback: exact FP32 WMMA (V_WMMA_F32_16X16X4_F32) ----------------
__global__ __launch_bounds__(256) void k_mmd(const float* __restrict__ src,
                                             const float* __restrict__ tgt,
                                             float* __restrict__ ws) {
    __shared__ float lm[BM * LDSW];
    __shared__ float ln[BN * LDSW];
    __shared__ float sqm[BM];
    __shared__ float sqn[BN];
    __shared__ float red[256];

    const int t    = threadIdx.x;
    const int lane = t & 31;
    const int wid  = t >> 5;
    const int wm   = wid >> 2;
    const int wn   = wid & 3;
    const int bi   = blockIdx.x;
    const int bj   = blockIdx.y;
    const int m0   = bi * BM;
    const int n0   = bj * BN;
    const int rl   = lane & 15;
    const int koff = (lane >> 4) << 1;
    const int mh   = (lane >> 4) << 3;

    if (t < BM) sqm[t] = ws[WS_SQ + m0 + t];
    else        sqn[t - BM] = ws[WS_SQ + n0 + (t - BM)];

    v8f acc[4][2];
    const v8f vzero = {0.f, 0.f, 0.f, 0.f, 0.f, 0.f, 0.f, 0.f};
#pragma unroll
    for (int mt = 0; mt < 4; ++mt)
#pragma unroll
        for (int nt = 0; nt < 2; ++nt) acc[mt][nt] = vzero;

    for (int k0 = 0; k0 < DIM; k0 += BK) {
#pragma unroll
        for (int q = 0; q < 4; ++q) {
            const int s  = t + 256 * q;
            const int r  = s >> 3;
            const int c4 = (s & 7) << 2;
            const float4 xm = *(const float4*)(row_ptr(src, tgt, m0 + r) + k0 + c4);
            *(float4*)&lm[r * LDSW + c4] = xm;
            const float4 xn = *(const float4*)(row_ptr(src, tgt, n0 + r) + k0 + c4);
            *(float4*)&ln[r * LDSW + c4] = xn;
        }
        __syncthreads();

#pragma unroll
        for (int kk = 0; kk < BK; kk += 4) {
            v2f a[4], b[2];
#pragma unroll
            for (int mt = 0; mt < 4; ++mt)
                a[mt] = *(const v2f*)&lm[(wm * 64 + mt * 16 + rl) * LDSW + kk + koff];
#pragma unroll
            for (int nt = 0; nt < 2; ++nt)
                b[nt] = *(const v2f*)&ln[(wn * 32 + nt * 16 + rl) * LDSW + kk + koff];
#pragma unroll
            for (int mt = 0; mt < 4; ++mt)
#pragma unroll
                for (int nt = 0; nt < 2; ++nt)
                    acc[mt][nt] = __builtin_amdgcn_wmma_f32_16x16x4_f32(
                        false, a[mt], false, b[nt], (short)0, acc[mt][nt], false, false);
        }
        __syncthreads();
    }

    const float nb0 = ws[WS_NIB + 0], nb1 = ws[WS_NIB + 1], nb2 = ws[WS_NIB + 2],
                nb3 = ws[WS_NIB + 3], nb4 = ws[WS_NIB + 4];

    float lsum = 0.0f;
#pragma unroll
    for (int mt = 0; mt < 4; ++mt) {
#pragma unroll
        for (int nt = 0; nt < 2; ++nt) {
            const float sj = sqn[wn * 32 + nt * 16 + rl];
#pragma unroll
            for (int v = 0; v < 8; ++v) {
                const float si = sqm[wm * 64 + mt * 16 + mh + v];
                float l2 = si + sj - 2.0f * acc[mt][nt][v];
                l2 = fmaxf(l2, 0.0f);
                lsum += __expf(l2 * nb0) + __expf(l2 * nb1) + __expf(l2 * nb2)
                      + __expf(l2 * nb3) + __expf(l2 * nb4);
            }
        }
    }

    red[t] = lsum;
    __syncthreads();
    for (int s = 128; s > 0; s >>= 1) {
        if (t < s) red[t] += red[t + s];
        __syncthreads();
    }
    if (t == 0) {
        const float sgn = ((bi < (NROWS / BM)) == (bj < (NROWS / BN))) ? 1.0f : -1.0f;
        atomicAdd((double*)&ws[WS_ACC], (double)(sgn * red[0]));
    }
}

__global__ void k_writeout(const float* __restrict__ ws, float* __restrict__ out) {
    if (threadIdx.x == 0) {
        double acc = *(const double*)&ws[WS_ACC];
        out[0] = (float)(acc / ((double)NROWS * (double)NROWS));
    }
}

extern "C" void kernel_launch(void* const* d_in, const int* in_sizes, int n_in,
                              void* d_out, int out_size, void* d_ws, size_t ws_size,
                              hipStream_t stream) {
    const float* src = (const float*)d_in[0];
    const float* tgt = (const float*)d_in[1];
    float* out = (float*)d_out;
    float* ws  = (float*)d_ws;

    k_zero<<<1, 512, 0, stream>>>(ws, out);
    k_rowstats<<<NTOT / 32, 256, 0, stream>>>(src, tgt, ws);
    k_finalize<<<1, 256, 0, stream>>>(ws);

    dim3 grid(NTOT / BM, NTOT / BN);
    const size_t bfelems = (size_t)NTOT * DIM;
    const size_t need = (size_t)WS_BF16_BYTE_OFF + 2 * bfelems * sizeof(unsigned short);
    if (ws_size >= need) {
        unsigned short* xhi = (unsigned short*)((char*)d_ws + WS_BF16_BYTE_OFF);
        unsigned short* xlo = xhi + bfelems;
        k_convert<<<(int)(bfelems / (8 * 256)), 256, 0, stream>>>(src, tgt, xhi, xlo);
        k_mmd_bf16<<<grid, 256, 0, stream>>>(xhi, xlo, ws);
    } else {
        k_mmd<<<grid, 256, 0, stream>>>(src, tgt, ws);
    }
    k_writeout<<<1, 64, 0, stream>>>(ws, out);
}
